// Point_TemporalPriorEncoder_6047313953088
// MI455X (gfx1250) — matrix operand order repaired and checked by reading, not compiled, not run on hardware
//
#include <hip/hip_runtime.h>

typedef __attribute__((ext_vector_type(16))) _Float16 v16h;
typedef __attribute__((ext_vector_type(8)))  _Float16 v8h;
typedef __attribute__((ext_vector_type(8)))  float    v8f;
typedef __attribute__((ext_vector_type(4)))  unsigned int u32x4;
typedef __attribute__((ext_vector_type(8)))  int          i32x8;
typedef __attribute__((ext_vector_type(4)))  int          i32x4;

#if defined(__has_builtin)
#if __has_builtin(__builtin_amdgcn_tensor_load_to_lds)
#define ATHENA_TDM 1
#endif
#endif

// ---------------------------------------------------------------------------
// Weight packing: f32 [tap][cin][cout] -> f16 fragments laid out exactly as the
// wave32 B-operand of v_wmma_f32_16x16x32_f16 expects:
//   bp[(((tap*KC+kc)*NT+nt)*32 + lane)*16 + e] = W[cin = kc*32 + (e>>3)*16 + (lane>>4)*8 + (e&7)]
//                                                 [cout = nt*16 + (lane&15)]
// (zero-padded when cin >= CIN, used for CIN==16 convs)
// ---------------------------------------------------------------------------
template<int CIN, int COUT, int TAPS>
__global__ void pack_w(const float* __restrict__ w, _Float16* __restrict__ bp)
{
  constexpr int KC = (CIN + 31) / 32, NT = COUT / 16;
  const int total = TAPS * KC * NT * 512;
  for (int idx = blockIdx.x * blockDim.x + threadIdx.x; idx < total;
       idx += gridDim.x * blockDim.x) {
    const int e    = idx & 15;
    const int lane = (idx >> 4) & 31;
    const int frag = idx >> 9;            // (tap*KC + kc)*NT + nt
    const int nt   = frag % NT;
    const int kc   = (frag / NT) % KC;
    const int tap  = frag / (NT * KC);
    const int cin  = kc * 32 + (e >> 3) * 16 + (lane >> 4) * 8 + (e & 7);
    const int cout = nt * 16 + (lane & 15);
    float v = (cin < CIN) ? w[((long)tap * CIN + cin) * COUT + cout] : 0.0f;
    bp[idx] = (_Float16)v;
  }
}

// ---------------------------------------------------------------------------
__global__ void scatter_fused(const int* __restrict__ coords,
                              const float* __restrict__ f1,
                              const float* __restrict__ f2,
                              _Float16* __restrict__ x64,
                              float* __restrict__ mask64, int N)
{
  int idx = blockIdx.x * blockDim.x + threadIdx.x;
  if (idx >= N * 128) return;
  const int n = idx >> 7, c = idx & 127;
  const int ci = coords[n * 3], cj = coords[n * 3 + 1], ck = coords[n * 3 + 2];
  const long site = ((long)ci * 64 + cj) * 64 + ck;
  float v = (c < 64) ? f1[(long)n * 64 + c] : f2[(long)n * 64 + (c - 64)];
  x64[site * 128 + c] = (_Float16)v;
  if (c == 0) mask64[site] = 1.0f;
}

__global__ void scatter_mask_ds(const int* __restrict__ cds,
                                float* __restrict__ mask32, int M)
{
  int i = blockIdx.x * blockDim.x + threadIdx.x;
  if (i >= M) return;
  const int di = cds[i * 3], dj = cds[i * 3 + 1], dk = cds[i * 3 + 2];
  mask32[((long)di * 32 + dj) * 32 + dk] = 1.0f;
}

__global__ void gather_out(const int* __restrict__ cds,
                           const _Float16* __restrict__ x,
                           float* __restrict__ out, int M)
{
  int idx = blockIdx.x * blockDim.x + threadIdx.x;
  if (idx >= M * 64) return;
  const int i = idx >> 6, c = idx & 63;
  const int di = cds[i * 3], dj = cds[i * 3 + 1], dk = cds[i * 3 + 2];
  out[idx] = (float)x[(((long)di * 32 + dj) * 32 + dk) * 64 + c];
}

// ---------------------------------------------------------------------------
// Dense-masked 3D conv as WMMA implicit GEMM (wave32, 16 sites x COUT / wave).
// LDSW: stage the entire packed-B array into LDS once per block via the
// Tensor Data Mover (tensor_load_to_lds + s_wait_tensorcnt), then serve all
// B fragments from LDS (ds_load_b128). Used when packed weights fit <=64KB.
// ---------------------------------------------------------------------------
template<int CIN, int COUT, int KD, int STRIDE, int DIMIN, int DIMOUT,
         bool RELU, bool RESADD, bool LDSW>
__global__ __launch_bounds__(128) void conv_wmma(
    const _Float16* __restrict__ in,
    _Float16* __restrict__ out,
    const _Float16* __restrict__ bp,
    const float* __restrict__ bias,
    const float* __restrict__ mask,
    const _Float16* __restrict__ res,
    int coutStride, int coutOfs)
{
  constexpr int KC     = (CIN + 31) / 32;
  constexpr int NT     = COUT / 16;
  constexpr int TAPS   = KD * KD * KD;
  constexpr int PAD    = (KD == 3) ? 1 : 0;
  constexpr int SMEM_H = LDSW ? TAPS * KC * NT * 512 : 16;  // halves

  __shared__ __align__(64) _Float16 smem[SMEM_H];

  if constexpr (LDSW) {
#if defined(ATHENA_TDM)
    if (threadIdx.x < 32) {  // one TDM issue from wave 0 (EXEC-independent DMA)
      constexpr unsigned nE = (unsigned)(SMEM_H / 4);  // 8-byte elements
      const unsigned long long ga = (unsigned long long)bp;
      const unsigned ldsOff = (unsigned)(unsigned long long)(&smem[0]);
      u32x4 g0;
      g0[0] = 1u;                                    // count=1 valid user D#
      g0[1] = ldsOff;                                // lds_addr (bytes)
      g0[2] = (unsigned)(ga & 0xFFFFFFFFu);          // global_addr[31:0]
      g0[3] = (unsigned)((ga >> 32) & 0x1FFFFFFu) | (2u << 30);  // [56:32]|type=2
      i32x8 g1;
      g1[0] = 3 << 16;                               // data_size = 8B
      g1[1] = (int)((nE & 0xFFFFu) << 16);           // tensor_dim0[15:0]
      g1[2] = (int)((nE >> 16) & 0xFFFFu) | (1 << 16);  // dim0 hi | tensor_dim1=1
      g1[3] = (int)((nE & 0xFFFFu) << 16);           // dim1 hi=0 | tile_dim0=nE
      g1[4] = 0;                                     // tile_dim1/2 unused
      g1[5] = (int)nE;                               // tensor_dim0_stride lo
      g1[6] = 0;                                     // stride hi | dim1_stride lo
      g1[7] = 0;
      i32x4 z4 = {0, 0, 0, 0};
#if __clang_major__ >= 23
      i32x8 z8 = {0, 0, 0, 0, 0, 0, 0, 0};
      __builtin_amdgcn_tensor_load_to_lds(g0, g1, z4, z4, z8, 0);
#else
      __builtin_amdgcn_tensor_load_to_lds(g0, g1, z4, z4, 0);
#endif
      __builtin_amdgcn_s_wait_tensorcnt(0);
    }
#else
    for (int i = threadIdx.x * 16; i < SMEM_H; i += 128 * 16)
      *(v16h*)(smem + i) = *(const v16h*)(bp + i);
#endif
    __syncthreads();
  }

  const int lane  = threadIdx.x & 31;
  const int wave  = threadIdx.x >> 5;
  const int tile  = blockIdx.x * 4 + wave;
  const int site0 = tile * 16;

  const int x0 = site0 % DIMOUT;                 // compile-time pow2 -> mask
  const int y  = (site0 / DIMOUT) % DIMOUT;
  const int z  = site0 / (DIMOUT * DIMOUT);

  const int M  = lane & 15;
  const int kh = lane >> 4;
  const int xm = x0 + M;

  const v8f zero8 = {};
  const v8h zeroh = {};
  v8f acc[NT];
#pragma unroll
  for (int n = 0; n < NT; ++n) acc[n] = zero8;

  for (int tap = 0; tap < TAPS; ++tap) {
    const int dz = tap / (KD * KD);
    const int dy = (tap / KD) % KD;
    const int dx = tap % KD;
    const int iz = z * STRIDE + dz - PAD;
    const int iy = y * STRIDE + dy - PAD;
    const int ix = xm * STRIDE + dx - PAD;
    const bool valid = ((unsigned)iz < (unsigned)DIMIN) &&
                       ((unsigned)iy < (unsigned)DIMIN) &&
                       ((unsigned)ix < (unsigned)DIMIN);
    const _Float16* arow = in + (((long)iz * DIMIN + iy) * DIMIN + ix) * CIN;

#pragma unroll
    for (int kc = 0; kc < KC; ++kc) {
      v8h lo = zeroh, hi = zeroh;
      const int c0 = kc * 32 + kh * 8;
      const int c1 = c0 + 16;
      if (valid) {
        lo = *(const v8h*)(arow + c0);        // K = kbase + kh*8 .. +7
        if (c1 < CIN)
          hi = *(const v8h*)(arow + c1);      // K = kbase + 16 + kh*8 .. +7
      }
      v16h a = __builtin_shufflevector(lo, hi, 0, 1, 2, 3, 4, 5, 6, 7,
                                       8, 9, 10, 11, 12, 13, 14, 15);
#pragma unroll
      for (int nt = 0; nt < NT; ++nt) {
        const long bofs = (long)((tap * KC + kc) * NT + nt) * 512 + lane * 16;
        v16h b;
        if constexpr (LDSW) b = *(const v16h*)(smem + bofs);   // ds_load
        else                b = *(const v16h*)(bp + bofs);     // global/L2
        acc[nt] = __builtin_amdgcn_wmma_f32_16x16x32_f16(
            false, a, false, b, (short)0, acc[nt], false, false);
      }
    }
  }

  // epilogue: bias -> mask -> (ReLU) -> (+residual) -> f16 store
  float m[8];
#pragma unroll
  for (int r = 0; r < 8; ++r) m[r] = mask[site0 + kh * 8 + r];

#pragma unroll
  for (int nt = 0; nt < NT; ++nt) {
    const int ch = nt * 16 + (lane & 15);
    const float bv = bias[ch];
#pragma unroll
    for (int r = 0; r < 8; ++r) {
      const long site = site0 + kh * 8 + r;
      float v = acc[nt][r] + bv;
      v *= m[r];
      if (RELU) v = v > 0.0f ? v : 0.0f;
      const long o = site * coutStride + coutOfs + ch;
      if (RESADD) v += (float)res[o];
      out[o] = (_Float16)v;
    }
  }
}

// ---------------------------------------------------------------------------
extern "C" void kernel_launch(void* const* d_in, const int* in_sizes, int n_in,
                              void* d_out, int out_size, void* d_ws,
                              size_t ws_size, hipStream_t stream)
{
  const int*   coords    = (const int*)d_in[0];
  const int*   coords_ds = (const int*)d_in[1];
  const float* feats_c1  = (const float*)d_in[2];
  const float* feats_ref = (const float*)d_in[3];
  const float* w_fuse  = (const float*)d_in[4];
  const float* b_fuse  = (const float*)d_in[5];
  const float* w_down  = (const float*)d_in[6];
  const float* b_down  = (const float*)d_in[7];
  const float* w_conv2 = (const float*)d_in[8];
  const float* b_conv2 = (const float*)d_in[9];
  const float* w_p00 = (const float*)d_in[10];
  const float* b_p00 = (const float*)d_in[11];
  const float* w_p01 = (const float*)d_in[12];
  const float* b_p01 = (const float*)d_in[13];
  const float* w_p10 = (const float*)d_in[14];
  const float* b_p10 = (const float*)d_in[15];
  const float* w_p11 = (const float*)d_in[16];
  const float* b_p11 = (const float*)d_in[17];
  const float* w_p12 = (const float*)d_in[18];
  const float* b_p12 = (const float*)d_in[19];

  const int N = in_sizes[0] / 3;
  const int M = in_sizes[1] / 3;
  const int L = in_sizes[10] / (27 * 64 * 16);

  // ---- workspace carve-up (f16 activation grids; ~113MB, fits MI455X L2)
  char*  ws = (char*)d_ws;
  size_t o  = 0;
  auto carve = [&](size_t bytes) { size_t c = o; o += (bytes + 255) & ~(size_t)255; return c; };

  _Float16* x64    = (_Float16*)(ws + carve((size_t)262144 * 128 * 2)); // 67 MB
  float*    mask64 = (float*)   (ws + carve((size_t)262144 * 4));
  float*    mask32 = (float*)   (ws + carve((size_t)32768 * 4));
  _Float16* g1     = (_Float16*)(ws + carve((size_t)262144 * 64 * 2));  // 33.5 MB
  _Float16* xa     = (_Float16*)(ws + carve((size_t)32768 * 64 * 2));
  _Float16* xb     = (_Float16*)(ws + carve((size_t)32768 * 64 * 2));
  _Float16* t16a   = (_Float16*)(ws + carve((size_t)32768 * 16 * 2));
  _Float16* t16b   = (_Float16*)(ws + carve((size_t)32768 * 16 * 2));

  const size_t szFuse  = 27 * 4 * 4 * 512;  // halves
  const size_t szDown  = 8 * 2 * 4 * 512;
  const size_t szConv2 = 27 * 2 * 4 * 512;
  const size_t szP00   = 27 * 2 * 1 * 512;
  const size_t szP01   = 27 * 1 * 2 * 512;
  const size_t szP10   = 1 * 2 * 1 * 512;
  const size_t szP11   = 27 * 1 * 1 * 512;
  const size_t szP12   = 1 * 1 * 2 * 512;
  _Float16* bpFuse  = (_Float16*)(ws + carve(szFuse * 2));
  _Float16* bpDown  = (_Float16*)(ws + carve(szDown * 2));
  _Float16* bpConv2 = (_Float16*)(ws + carve(szConv2 * 2));
  _Float16* bpP00   = (_Float16*)(ws + carve(szP00 * 2 * L));
  _Float16* bpP01   = (_Float16*)(ws + carve(szP01 * 2 * L));
  _Float16* bpP10   = (_Float16*)(ws + carve(szP10 * 2 * L));
  _Float16* bpP11   = (_Float16*)(ws + carve(szP11 * 2 * L));
  _Float16* bpP12   = (_Float16*)(ws + carve(szP12 * 2 * L));
  if (o > ws_size) return;  // workspace too small

  // ---- zero-init the scatter targets (re-done every call; graph-safe)
  hipMemsetAsync(x64, 0, (size_t)262144 * 128 * 2, stream);
  hipMemsetAsync(mask64, 0, (size_t)262144 * 4, stream);
  hipMemsetAsync(mask32, 0, (size_t)32768 * 4, stream);

  // ---- pack weights into WMMA B-fragment layout
  auto pg = [](size_t halves) { return dim3((unsigned)((halves + 255) / 256)); };
  pack_w<128, 64, 27><<<pg(szFuse), 256, 0, stream>>>(w_fuse, bpFuse);
  pack_w<64, 64, 8><<<pg(szDown), 256, 0, stream>>>(w_down, bpDown);
  pack_w<64, 64, 27><<<pg(szConv2), 256, 0, stream>>>(w_conv2, bpConv2);
  for (int i = 0; i < L; ++i) {
    pack_w<64, 16, 27><<<pg(szP00), 256, 0, stream>>>(w_p00 + (size_t)i * 27 * 64 * 16, bpP00 + (size_t)i * szP00);
    pack_w<16, 32, 27><<<pg(szP01), 256, 0, stream>>>(w_p01 + (size_t)i * 27 * 16 * 32, bpP01 + (size_t)i * szP01);
    pack_w<64, 16, 1><<<pg(szP10), 256, 0, stream>>>(w_p10 + (size_t)i * 64 * 16, bpP10 + (size_t)i * szP10);
    pack_w<16, 16, 27><<<pg(szP11), 256, 0, stream>>>(w_p11 + (size_t)i * 27 * 16 * 16, bpP11 + (size_t)i * szP11);
    pack_w<16, 32, 1><<<pg(szP12), 256, 0, stream>>>(w_p12 + (size_t)i * 16 * 32, bpP12 + (size_t)i * szP12);
  }

  // ---- scatter inputs
  scatter_fused<<<(N * 128 + 255) / 256, 256, 0, stream>>>(coords, feats_c1, feats_ref, x64, mask64, N);
  scatter_mask_ds<<<(M + 255) / 256, 256, 0, stream>>>(coords_ds, mask32, M);

  // ---- network
  // fuse: 3^3, 128->64 over 64^3 (dominant GEMM; B stays in L2, too big for LDS)
  conv_wmma<128, 64, 3, 1, 64, 64, true, false, false><<<4096, 128, 0, stream>>>(
      x64, g1, bpFuse, b_fuse, mask64, nullptr, 64, 0);
  // down: 2^3 stride-2, 64->64 -> 32^3 (B via TDM -> LDS)
  conv_wmma<64, 64, 2, 2, 64, 32, true, false, true><<<512, 128, 0, stream>>>(
      g1, xa, bpDown, b_down, mask32, nullptr, 64, 0);
  // conv2: 3^3, 64->64 (no relu; 216KB packed B -> keep in L2)
  conv_wmma<64, 64, 3, 1, 32, 32, false, false, false><<<512, 128, 0, stream>>>(
      xa, xb, bpConv2, b_conv2, mask32, nullptr, 64, 0);

  _Float16* cur = xb;
  _Float16* nxt = xa;
  for (int i = 0; i < L; ++i) {
    // branch 0: 3^3 64->16 relu ; 3^3 16->32 (+residual into channels 0..31)
    conv_wmma<64, 16, 3, 1, 32, 32, true, false, true><<<512, 128, 0, stream>>>(
        cur, t16a, bpP00 + (size_t)i * szP00, b_p00 + i * 16, mask32, nullptr, 16, 0);
    conv_wmma<16, 32, 3, 1, 32, 32, false, true, true><<<512, 128, 0, stream>>>(
        t16a, nxt, bpP01 + (size_t)i * szP01, b_p01 + i * 32, mask32, cur, 64, 0);
    // branch 1: 1^3 64->16 relu ; 3^3 16->16 relu ; 1^3 16->32 (+res ch 32..63)
    conv_wmma<64, 16, 1, 1, 32, 32, true, false, true><<<512, 128, 0, stream>>>(
        cur, t16b, bpP10 + (size_t)i * szP10, b_p10 + i * 16, mask32, nullptr, 16, 0);
    conv_wmma<16, 16, 3, 1, 32, 32, true, false, true><<<512, 128, 0, stream>>>(
        t16b, t16a, bpP11 + (size_t)i * szP11, b_p11 + i * 16, mask32, nullptr, 16, 0);
    conv_wmma<16, 32, 1, 1, 32, 32, false, true, true><<<512, 128, 0, stream>>>(
        t16a, nxt, bpP12 + (size_t)i * szP12, b_p12 + i * 32, mask32, cur, 64, 32);
    _Float16* tmp = cur; cur = nxt; nxt = tmp;
  }

  // ---- gather output at downsampled coords (f32 out)
  gather_out<<<(M * 64 + 255) / 256, 256, 0, stream>>>(coords_ds, cur, (float*)d_out, M);
}
